// UnifiedMatcherMoreThanOne_38001870635897
// MI455X (gfx1250) — matrix operand order repaired and checked by reading, not compiled, not run on hardware
//
#include <hip/hip_runtime.h>
#include <hip/hip_bf16.h>

typedef __attribute__((ext_vector_type(16))) _Float16 v16h;
typedef __attribute__((ext_vector_type(8)))  float    v8f;
typedef __attribute__((ext_vector_type(8)))  unsigned v8u;

#define N_PRED 65536
#define N_GT   128
#define N_CLS  256
#define IOU_THRES 0.25f
#define MAX_DYN 10
#define EPSV 1e-6f
#define NEGV -1.0e9f
#define HT 1024

__device__ __forceinline__ float sigmoidf(float x) { return 1.0f / (1.0f + __expf(-x)); }

// ---------------------------------------------------------------------------
// Kernel 1: fused cost_class (WMMA one-hot GEMM on raw logits) + 3D GIoU.
//   grid = N_PRED/16 blocks, 256 threads = 8 waves; wave w owns N-tile w*16.
//   - Stage all_cls[16x256] f32->f16 in LDS (coalesced float4 loads, pure
//     packed converts; sigmoid applied AFTER the gather-GEMM since the
//     one-hot selects a single element: sigmoid(gather) == gather(sigmoid)).
//   - 8x V_WMMA_F32_16X16X32_F16 fully unrolled; one-hot B built with a
//     sparse trick: only one K-step fragment per lane is nonzero.
//   - GIoU fused per C element; cost tile transposed via LDS for bursts.
// ---------------------------------------------------------------------------
__global__ __launch_bounds__(256)
void cost_giou_kernel(const float* __restrict__ all_centers,
                      const float* __restrict__ all_sizes,
                      const float* __restrict__ all_cls,
                      const float* __restrict__ gt_centers,
                      const float* __restrict__ gt_sizes,
                      const long long* __restrict__ gt_labels,
                      float* __restrict__ giou_mat,   // [N_PRED][N_GT]
                      float* __restrict__ costT)      // [N_GT][N_PRED]
{
    __shared__ _Float16 s_a[16 * N_CLS];     // 8 KB: logit tile, f16
    __shared__ float    s_cost[N_GT * 16];   // 8 KB: cost tile, gt-major
    __shared__ float    s_pc[16 * 3], s_ps[16 * 3];
    __shared__ float    s_gc[N_GT * 3], s_gs[N_GT * 3];

    const int tid   = threadIdx.x;
    const int mtile = blockIdx.x * 16;

    // ---- stage boxes ----
    for (int t = tid; t < 48; t += 256) {
        s_pc[t] = all_centers[mtile * 3 + t];
        s_ps[t] = all_sizes[mtile * 3 + t];
    }
    for (int t = tid; t < N_GT * 3; t += 256) {
        s_gc[t] = gt_centers[t];
        s_gs[t] = gt_sizes[t];
    }

    // ---- stage A tile: 16x256 f32 -> f16 in LDS (coalesced, pk converts) --
    {
        const float4* src = (const float4*)(all_cls + (size_t)mtile * N_CLS);
#pragma unroll
        for (int i = 0; i < 4; ++i) {
            const int f = tid + i * 256;          // float4 index, 1024 total
            const float4 q = src[f];
            const int e = f * 4;
            s_a[e + 0] = (_Float16)q.x;
            s_a[e + 1] = (_Float16)q.y;
            s_a[e + 2] = (_Float16)q.z;
            s_a[e + 3] = (_Float16)q.w;
        }
    }
    // prefetch next block's A tile into GL2 (global_prefetch_b8)
    if (tid == 0 && blockIdx.x + 1 < (int)gridDim.x)
        __builtin_prefetch(all_cls + (size_t)(mtile + 16) * N_CLS, 0, 1);
    __syncthreads();

    const int wave = tid >> 5;
    const int lane = tid & 31;
    const int h    = lane >> 4;      // half-wave (16x16 layout split)
    const int l15  = lane & 15;
    const int n0   = wave * 16;      // this wave's N-tile (8 waves x 16 = 128)
    const int nG   = n0 + l15;       // gt column owned by this lane (B/C frag)

    // ---- sparse one-hot B precompute: exactly one nonzero fragment ----
    const int      label  = (int)gt_labels[nG];
    const bool     mine   = (((label >> 3) & 1) == h);  // my half owns it?
    const int      ktstar = label >> 5;                 // which K-step
    const int      jstar  = ((label >> 1) & 3) | (((label >> 4) & 1) << 2); // which dword
    const unsigned hword  = 0x3C00u << ((label & 1) * 16);
    unsigned mask8[8];
#pragma unroll
    for (int d = 0; d < 8; ++d)
        mask8[d] = (mine && jstar == d) ? hword : 0u;

    const _Float16* arow = s_a + l15 * N_CLS;   // A-frag row for this lane

    v8f c = {};
#pragma unroll
    for (int kt = 0; kt < N_CLS / 32; ++kt) {
        // A fragment (16-bit A 16x32 layout):
        // VGPR v holds K = kt*32 + (v>=4?16:0) + h*8 + (v&3)*2 .. +1
        v16h a;
#pragma unroll
        for (int v = 0; v < 8; ++v) {
            const int kb = kt * 32 + ((v >= 4) ? 16 : 0) + h * 8 + (v & 3) * 2;
            a[2 * v + 0] = arow[kb];
            a[2 * v + 1] = arow[kb + 1];
        }
        // B fragment: zero except when kt == ktstar (1 cmp + 8 selects)
        v8u bw;
#pragma unroll
        for (int d = 0; d < 8; ++d)
            bw[d] = (kt == ktstar) ? mask8[d] : 0u;
        const v16h b = __builtin_bit_cast(v16h, bw);

        c = __builtin_amdgcn_wmma_f32_16x16x32_f16(false, a, false, b,
                                                   (short)0, c, false, false);
    }

    // ---- fused GIoU + total cost per C element ----
    float glo[3], ghi[3], gvol = 1.0f;
#pragma unroll
    for (int d = 0; d < 3; ++d) {
        const float cg = s_gc[nG * 3 + d], sg = s_gs[nG * 3 + d];
        glo[d] = cg - 0.5f * sg;
        ghi[d] = cg + 0.5f * sg;
        gvol *= sg;
    }

    // C layout: VGPR r -> M = r + 8*h (lanes 0-15: M=r, lanes 16-31: M=r+8)
#pragma unroll
    for (int r = 0; r < 8; ++r) {
        const int mloc = r + 8 * h;
        const int m    = mtile + mloc;
        float pvol = 1.0f, ov = 1.0f, en = 1.0f;
#pragma unroll
        for (int d = 0; d < 3; ++d) {
            const float cp = s_pc[mloc * 3 + d], sp = s_ps[mloc * 3 + d];
            const float lo = cp - 0.5f * sp, hi = cp + 0.5f * sp;
            pvol *= sp;
            ov *= fmaxf(fminf(hi, ghi[d]) - fmaxf(lo, glo[d]), 0.0f);
            en *= fmaxf(fmaxf(hi, ghi[d]) - fminf(lo, glo[d]), 0.0f);
        }
        const float uni  = fmaxf(pvol + gvol - ov, EPSV);
        const float iou  = ov / uni;
        const float enc  = fmaxf(en, EPSV);
        const float giou = iou - (enc - uni) / enc;

        giou_mat[(size_t)m * N_GT + nG] = giou;     // 64B bursts per 16 lanes
        // total = -sigmoid(logit at label) - 2*giou; transpose via LDS
        s_cost[nG * 16 + mloc] = -sigmoidf(c[r]) - 2.0f * giou;
    }
    __syncthreads();

    // write cost tile gt-major with contiguous 16-float bursts per gt row
    for (int idx = tid; idx < N_GT * 16; idx += 256) {
        const int g    = idx >> 4;
        const int mloc = idx & 15;
        costT[(size_t)g * N_PRED + mtile + mloc] = s_cost[idx];
    }
}

// ---------------------------------------------------------------------------
// Kernel 2: per-gt column max of GIoU (128 blocks x 256 threads)
// ---------------------------------------------------------------------------
__global__ __launch_bounds__(256)
void colmax_kernel(const float* __restrict__ giou_mat, float* __restrict__ maxg)
{
    __shared__ float red[256];
    const int g = blockIdx.x, tid = threadIdx.x;
    float m = -__builtin_inff();
    for (int p = tid; p < N_PRED; p += 256)
        m = fmaxf(m, giou_mat[(size_t)p * N_GT + g]);
    red[tid] = m;
    __syncthreads();
    for (int s = 128; s > 0; s >>= 1) {
        if (tid < s) red[tid] = fmaxf(red[tid], red[tid + s]);
        __syncthreads();
    }
    if (tid == 0) maxg[g] = red[0];
}

// ---------------------------------------------------------------------------
// Kernel 3: stable ascending argsort of 128 values by rank counting
// ---------------------------------------------------------------------------
__global__ void argsort128_kernel(const float* __restrict__ maxg,
                                  int* __restrict__ order)
{
    const int g = threadIdx.x;
    if (g >= N_GT) return;
    const float mv = maxg[g];
    int rank = 0;
    for (int j = 0; j < N_GT; ++j) {
        const float o = maxg[j];
        rank += (o < mv) || (o == mv && j < g);
    }
    order[rank] = g;
}

// ---------------------------------------------------------------------------
// Kernel 4: Jonker-Volgenant augmenting-path LSA on costT (128 x 65536).
//   Single 1024-thread workgroup; 65536-wide relax/argmin passes are
//   L2-resident (32 MB costT << 192 MB L2).
// ---------------------------------------------------------------------------
__global__ __launch_bounds__(HT)
void hungarian_kernel(const float* __restrict__ costT,
                      float* __restrict__ u, float* __restrict__ v,
                      float* __restrict__ shortest, int* __restrict__ path,
                      int* __restrict__ SC, int* __restrict__ row4col,
                      int* __restrict__ col4row)
{
    __shared__ float red_v[HT];
    __shared__ int   red_i[HT];
    __shared__ int   sSR[N_GT];
    __shared__ int   s_i, s_sink, s_nSR;
    __shared__ float s_minval;
    const int tid = threadIdx.x;

    for (int j = tid; j < N_PRED; j += HT) { v[j] = 0.0f; row4col[j] = -1; }
    for (int g = tid; g < N_GT; g += HT)   { u[g] = 0.0f; col4row[g] = -1; }
    __syncthreads();

    for (int cur = 0; cur < N_GT; ++cur) {
        for (int j = tid; j < N_PRED; j += HT) {
            shortest[j] = __builtin_inff();
            path[j] = -1;
            SC[j] = 0;
        }
        if (tid == 0) { s_i = cur; s_minval = 0.0f; s_sink = -1; s_nSR = 0; }
        __syncthreads();

        while (true) {
            const int   i  = s_i;
            const float mv = s_minval;
            if (tid == 0) { sSR[s_nSR] = i; s_nSR++; }
            const float ui = u[i];
            const float* crow = costT + (size_t)i * N_PRED;

            // relax: shortest = min(shortest, mv + cost - u[i] - v)
            for (int j = tid; j < N_PRED; j += HT) {
                if (!SC[j]) {
                    const float r = mv + crow[j] - ui - v[j];
                    if (r < shortest[j]) { shortest[j] = r; path[j] = i; }
                }
            }
            __syncthreads();

            // argmin over non-scanned columns
            float bv = __builtin_inff();
            int   bi = -1;
            for (int j = tid; j < N_PRED; j += HT) {
                if (!SC[j]) {
                    const float sv = shortest[j];
                    if (sv < bv) { bv = sv; bi = j; }
                }
            }
            red_v[tid] = bv; red_i[tid] = bi;
            __syncthreads();
            for (int s = HT / 2; s > 0; s >>= 1) {
                if (tid < s) {
                    const float v2 = red_v[tid + s];
                    const int   i2 = red_i[tid + s];
                    if (i2 != -1 &&
                        (v2 < red_v[tid] ||
                         (v2 == red_v[tid] && (red_i[tid] == -1 || i2 < red_i[tid])))) {
                        red_v[tid] = v2; red_i[tid] = i2;
                    }
                }
                __syncthreads();
            }
            if (tid == 0) {
                const int bj = red_i[0];
                s_minval = red_v[0];
                SC[bj] = 1;
                const int r4c = row4col[bj];
                if (r4c == -1) s_sink = bj;
                else           s_i = r4c;
            }
            __syncthreads();
            if (s_sink != -1) break;
        }

        // dual variable updates (col4row read BEFORE augmentation, as in ref)
        const float mval = s_minval;
        const int   nSR  = s_nSR;
        if (tid < nSR) {
            const int i2 = sSR[tid];
            if (i2 == cur) u[cur] += mval;
            else           u[i2]  += mval - shortest[col4row[i2]];
        }
        for (int j = tid; j < N_PRED; j += HT)
            if (SC[j]) v[j] -= mval - shortest[j];
        __syncthreads();

        // augment along parent pointers
        if (tid == 0) {
            int j = s_sink;
            while (true) {
                const int i2 = path[j];
                row4col[j] = i2;
                const int tmp = col4row[i2];
                col4row[i2] = j;
                j = tmp;
                if (i2 == cur) break;
            }
        }
        __syncthreads();
    }
}

// ---------------------------------------------------------------------------
// Kernel 5: dynamic assignment — for each gt (in `order`), claim up to 10
// unused preds with giou > 0.25, by repeated argmax; emit int64 index pairs.
// ---------------------------------------------------------------------------
__global__ __launch_bounds__(HT)
void dyn_assign_kernel(const float* __restrict__ giou_mat,
                       const int* __restrict__ col4row,
                       const int* __restrict__ order,
                       int* __restrict__ used,
                       long long* __restrict__ out)  // [2*1408]
{
    __shared__ float red_v[HT];
    __shared__ int   red_i[HT];
    const int tid = threadIdx.x;
    const int TOT = N_GT * (1 + MAX_DYN);   // 1408

    for (int p = tid; p < N_PRED; p += HT) used[p] = 0;
    __syncthreads();
    if (tid < N_GT) {
        const int p = col4row[tid];
        used[p] = 1;
        out[tid]       = (long long)p;    // combined_preds[0:128]
        out[TOT + tid] = (long long)tid;  // combined_gts[0:128]
    }
    __syncthreads();

    for (int k = 0; k < N_GT; ++k) {
        const int g = order[k];
        for (int t = 0; t < MAX_DYN; ++t) {
            float bv = NEGV;
            int   bi = -1;
            for (int p = tid; p < N_PRED; p += HT) {
                const float gi = giou_mat[(size_t)p * N_GT + g];
                if (gi > IOU_THRES && !used[p] && gi > bv) { bv = gi; bi = p; }
            }
            red_v[tid] = bv; red_i[tid] = bi;
            __syncthreads();
            for (int s = HT / 2; s > 0; s >>= 1) {
                if (tid < s) {
                    const float v2 = red_v[tid + s];
                    const int   i2 = red_i[tid + s];
                    if (i2 != -1 &&
                        (v2 > red_v[tid] ||
                         (v2 == red_v[tid] && (red_i[tid] == -1 || i2 < red_i[tid])))) {
                        red_v[tid] = v2; red_i[tid] = i2;
                    }
                }
                __syncthreads();
            }
            if (tid == 0) {
                const int slot = N_GT + k * MAX_DYN + t;
                if (red_i[0] != -1) {
                    used[red_i[0]] = 1;
                    out[slot]       = (long long)red_i[0];
                    out[TOT + slot] = (long long)g;
                } else {
                    out[slot]       = -1ll;
                    out[TOT + slot] = -1ll;
                }
            }
            __syncthreads();
        }
    }
}

// ---------------------------------------------------------------------------
extern "C" void kernel_launch(void* const* d_in, const int* in_sizes, int n_in,
                              void* d_out, int out_size, void* d_ws, size_t ws_size,
                              hipStream_t stream)
{
    const float*     all_centers = (const float*)d_in[0];
    const float*     all_sizes   = (const float*)d_in[1];
    const float*     all_cls     = (const float*)d_in[2];
    const float*     all_objness = (const float*)d_in[3]; // weight 0 -> unused
    const float*     gt_centers  = (const float*)d_in[4];
    const float*     gt_sizes   = (const float*)d_in[5];
    const long long* gt_labels   = (const long long*)d_in[6];
    (void)all_objness; (void)in_sizes; (void)n_in; (void)out_size; (void)ws_size;

    char*  ws  = (char*)d_ws;
    size_t off = 0;
    auto alloc = [&](size_t bytes) -> void* {
        void* p = ws + off;
        off += (bytes + 255) & ~(size_t)255;
        return p;
    };

    float* giou_mat = (float*)alloc((size_t)N_PRED * N_GT * sizeof(float)); // 32 MB
    float* costT    = (float*)alloc((size_t)N_GT * N_PRED * sizeof(float)); // 32 MB
    float* v_dual   = (float*)alloc((size_t)N_PRED * sizeof(float));
    float* shortest = (float*)alloc((size_t)N_PRED * sizeof(float));
    int*   path     = (int*)  alloc((size_t)N_PRED * sizeof(int));
    int*   SC       = (int*)  alloc((size_t)N_PRED * sizeof(int));
    int*   row4col  = (int*)  alloc((size_t)N_PRED * sizeof(int));
    int*   used     = (int*)  alloc((size_t)N_PRED * sizeof(int));
    float* u_dual   = (float*)alloc((size_t)N_GT * sizeof(float));
    int*   col4row  = (int*)  alloc((size_t)N_GT * sizeof(int));
    float* maxg     = (float*)alloc((size_t)N_GT * sizeof(float));
    int*   order    = (int*)  alloc((size_t)N_GT * sizeof(int));

    cost_giou_kernel<<<N_PRED / 16, 256, 0, stream>>>(
        all_centers, all_sizes, all_cls, gt_centers, gt_sizes, gt_labels,
        giou_mat, costT);

    colmax_kernel<<<N_GT, 256, 0, stream>>>(giou_mat, maxg);

    argsort128_kernel<<<1, 128, 0, stream>>>(maxg, order);

    hungarian_kernel<<<1, HT, 0, stream>>>(
        costT, u_dual, v_dual, shortest, path, SC, row4col, col4row);

    dyn_assign_kernel<<<1, HT, 0, stream>>>(
        giou_mat, col4row, order, used, (long long*)d_out);
}
